// CustomAttn_Kernel_49916109914149
// MI455X (gfx1250) — compile-verified, matrix operand order checked by reading, not run on hardware
//
#include <hip/hip_runtime.h>

typedef __attribute__((ext_vector_type(16))) _Float16 v16h;
typedef __attribute__((ext_vector_type(8)))  _Float16 v8h;
typedef __attribute__((ext_vector_type(8)))  float    v8f;

#define BB 4
#define DD 64
#define LL 4096

// ---------------------------------------------------------------------------
// Kernel 1: x = pe^T @ W^T + b ; xs = x / lengthscale (stored f16 row-major
// [B][L][64]) ; sq[b][l] = ||xs||^2 (fp32).  Tiny vs. the Gram kernel.
// ---------------------------------------------------------------------------
__global__ __launch_bounds__(256) void rbf_proj_kernel(
    const float* __restrict__ pe,    // [B][D][L]
    const float* __restrict__ W,     // [D][D]  (out, in)
    const float* __restrict__ bias,  // [D]
    const float* __restrict__ lsp,   // scalar lengthscale
    _Float16* __restrict__ xs,       // [B][L][D] f16
    float* __restrict__ sq)          // [B][L]
{
  __shared__ float sW[64 * 64];      // sW[d*64 + k]
  __shared__ float sPE[64 * 64];     // sPE[k*64 + p]  (k = feature, p = pos)
  __shared__ float sPart[4][64];

  const int t   = threadIdx.x;
  const int blk = blockIdx.x;        // B * (L/64) blocks
  const int b   = blk >> 6;          // L/64 == 64
  const int l0  = (blk & 63) << 6;
  const float inv_ls = 1.0f / lsp[0];

#pragma unroll
  for (int it = 0; it < 16; ++it) {
    int idx = t + it * 256;
    int r = idx >> 6, c = idx & 63;
    sW[idx]  = W[idx];
    sPE[idx] = pe[(size_t)b * DD * LL + (size_t)r * LL + l0 + c];
  }
  __syncthreads();

  const int p     = t & 63;
  const int dg    = t >> 6;
  const int dbase = dg * 16;

  float acc[16];
#pragma unroll
  for (int i = 0; i < 16; ++i) acc[i] = bias[dbase + i];

  for (int k = 0; k < 64; ++k) {
    float pv = sPE[k * 64 + p];
#pragma unroll
    for (int i = 0; i < 16; ++i)
      acc[i] = __builtin_fmaf(sW[(dbase + i) * 64 + k], pv, acc[i]);
  }

  float part = 0.0f;
  v16h xh;
#pragma unroll
  for (int i = 0; i < 16; ++i) {
    float v = acc[i] * inv_ls;
    part = __builtin_fmaf(v, v, part);
    xh[i] = (_Float16)v;
  }
  *(v16h*)(xs + ((size_t)(b * LL + l0 + p)) * DD + dbase) = xh;

  sPart[dg][p] = part;
  __syncthreads();
  if (t < 64)
    sq[(size_t)b * LL + l0 + t] =
        sPart[0][t] + sPart[1][t] + sPart[2][t] + sPart[3][t];
}

// ---------------------------------------------------------------------------
// Kernel 2: out[b,i,j] = os * exp(-0.5 * max(sq_i + sq_j - 2*xs_i.xs_j, 0))
// One wave per 16x16 tile; 4 j-tiles per wave reuse the A fragment.
// D=64 -> two v_wmma_f32_16x16x32_f16 per tile.
// ---------------------------------------------------------------------------
__global__ __launch_bounds__(256) void rbf_gram_kernel(
    const _Float16* __restrict__ xs,  // [B][L][64] f16
    const float* __restrict__ sq,     // [B][L]
    const float* __restrict__ osp,    // scalar outputscale
    float* __restrict__ out)          // [B][L][L]
{
  const int lane = threadIdx.x & 31;
  const int wid  = (blockIdx.x * 256 + threadIdx.x) >> 5;
  // wave jobs: b(4) * ti(256) * jblk(64, each = 4 j-tiles)
  const int jblk = wid & 63;
  const int ti   = (wid >> 6) & 255;
  const int b    = wid >> 14;
  const int i0   = ti << 4;

  const float os   = osp[0];
  const int   row  = lane & 15;
  const int   msel = lane >> 4;   // 0: K 0-7/16-23, M 0-7 of C ; 1: K 8-15/24-31, M 8-15
  const int   koff = msel << 3;

  // A fragments: lane holds row (i0+row) of xs; identical pattern serves B.
  const _Float16* pA = xs + ((size_t)(b * LL) + i0 + row) * DD;
  v16h a0, a1;
  {
    v8h alo0 = *(const v8h*)(pA + koff);
    v8h ahi0 = *(const v8h*)(pA + 16 + koff);
    v8h alo1 = *(const v8h*)(pA + 32 + koff);
    v8h ahi1 = *(const v8h*)(pA + 48 + koff);
#pragma unroll
    for (int i = 0; i < 8; ++i) {
      a0[i] = alo0[i]; a0[8 + i] = ahi0[i];
      a1[i] = alo1[i]; a1[8 + i] = ahi1[i];
    }
  }
  // sq_i for this lane's 8 C rows (M = r + 8*msel)
  v8f sqi = *(const v8f*)(sq + (size_t)b * LL + i0 + msel * 8);

#pragma unroll
  for (int jj = 0; jj < 4; ++jj) {
    const int j0 = (((jblk << 2) + jj) << 4);
    const _Float16* pB = xs + ((size_t)(b * LL) + j0 + row) * DD;
    v16h b0, b1;
    {
      v8h blo0 = *(const v8h*)(pB + koff);
      v8h bhi0 = *(const v8h*)(pB + 16 + koff);
      v8h blo1 = *(const v8h*)(pB + 32 + koff);
      v8h bhi1 = *(const v8h*)(pB + 48 + koff);
#pragma unroll
      for (int i = 0; i < 8; ++i) {
        b0[i] = blo0[i]; b0[8 + i] = bhi0[i];
        b1[i] = blo1[i]; b1[8 + i] = bhi1[i];
      }
    }
    v8f c = {};
    c = __builtin_amdgcn_wmma_f32_16x16x32_f16(false, a0, false, b0,
                                               (short)0, c, false, false);
    c = __builtin_amdgcn_wmma_f32_16x16x32_f16(false, a1, false, b1,
                                               (short)0, c, false, false);

    const float sqj = sq[(size_t)b * LL + j0 + row];
    float* po = out + (size_t)b * LL * LL + (size_t)(i0 + msel * 8) * LL +
                j0 + row;
#pragma unroll
    for (int r = 0; r < 8; ++r) {
      float d = sqi[r] + sqj - 2.0f * c[r];
      d = fmaxf(d, 0.0f);
      po[(size_t)r * LL] = os * __expf(-0.5f * d);
    }
  }
}

extern "C" void kernel_launch(void* const* d_in, const int* in_sizes, int n_in,
                              void* d_out, int out_size, void* d_ws,
                              size_t ws_size, hipStream_t stream) {
  const float* pe   = (const float*)d_in[0];  // [4,64,4096]
  const float* W    = (const float*)d_in[1];  // [64,64]
  const float* bias = (const float*)d_in[2];  // [64]
  const float* ls   = (const float*)d_in[3];  // scalar
  const float* os   = (const float*)d_in[4];  // scalar
  float* out = (float*)d_out;                 // [4,4096,4096]

  _Float16* xs = (_Float16*)d_ws;                                  // 2 MB
  float* sq = (float*)((char*)d_ws +
                       (size_t)BB * LL * DD * sizeof(_Float16));   // +64 KB

  rbf_proj_kernel<<<BB * (LL / 64), 256, 0, stream>>>(pe, W, bias, ls, xs, sq);

  // waves = 4 * 256 * 64 = 65536 ; 8 waves per 256-thread block
  rbf_gram_kernel<<<(BB * 256 * 64) / 8, 256, 0, stream>>>(xs, sq, os, out);
}
